// H2OLlamaAttention_10926396801278
// MI455X (gfx1250) — matrix operand order repaired and checked by reading, not compiled, not run on hardware
//
#include <hip/hip_runtime.h>
#include <hip/hip_bf16.h>
#include <stdint.h>

typedef __attribute__((ext_vector_type(16))) _Float16 v16h;
typedef __attribute__((ext_vector_type(8)))  _Float16 v8h;   // 16 bytes
typedef __attribute__((ext_vector_type(4)))  _Float16 v4h;   // 8 bytes
typedef __attribute__((ext_vector_type(8)))  float    v8f;
typedef __attribute__((ext_vector_type(4)))  unsigned int u32x4;
typedef __attribute__((ext_vector_type(4)))  int      i32x4;
typedef __attribute__((ext_vector_type(8)))  int      i32x8;

constexpr int Sn = 2048, Dn = 4096, Hn = 32, KVHn = 8, HDn = 128;
constexpr int HHn = 128, RECENTn = 512, NREP = Hn / KVHn;      // 4
constexpr int NKEEP = HHn + RECENTn;                            // 640
constexpr int SELN = Sn - RECENTn;                              // 1536

#define WMMA_F16(a, b, c) \
  __builtin_amdgcn_wmma_f32_16x16x32_f16(false, (a), false, (b), (short)0, (c), false, false)

// --------------------------------------------------------------------------
// CDNA5 async global->LDS copy (ASYNCcnt path, ISA 08_async_tensor §4).
// LDS offset = low 32 bits of the flat shared pointer (aperture in [63:32]).
// --------------------------------------------------------------------------
__device__ __forceinline__ void async_copy_b128(void* lds_ptr, const void* g) {
  uint32_t lds = (uint32_t)(uintptr_t)lds_ptr;
  uint64_t ga  = (uint64_t)(uintptr_t)g;
  asm volatile("global_load_async_to_lds_b128 %0, %1, off"
               :: "v"(lds), "v"(ga) : "memory");
}
__device__ __forceinline__ void wait_async_zero() {
  asm volatile("s_wait_asynccnt 0" ::: "memory");
}

// --------------------------------------------------------------------------
// Tensor Data Mover: 2-D f16 tile load, global -> LDS, with LDS row padding.
// D# layout per CDNA5 ISA. rows x cols_halves tile,
// row stride = cols_halves elements, data_size = 2B.
// pad_i: DWORDs-before-pad code (5 => 64 DW = 256B). pad_a: pad code (3 => 4 DW).
// Issue from ONE wave; EXEC is ignored by TDM. Wait with s_wait_tensorcnt.
// --------------------------------------------------------------------------
__device__ __forceinline__ void tdm_load_tile_f16(void* lds_ptr, const void* gptr,
                                                  int rows, int cols_halves,
                                                  int pad_i, int pad_a) {
  uint64_t ga  = (uint64_t)(uintptr_t)gptr;
  uint32_t lds = (uint32_t)(uintptr_t)lds_ptr;
  u32x4 g0;
  g0[0] = 1u;                                               // count=1 (valid user D#)
  g0[1] = lds;                                              // lds_addr
  g0[2] = (uint32_t)ga;                                     // global_addr[31:0]
  g0[3] = (uint32_t)((ga >> 32) & 0x1FFFFFFu) | (2u << 30); // global_addr[56:32], type=2
  i32x8 g1;
  uint32_t d0 = (1u << 16)                                  // data_size = 2B
              | (1u << 20)                                  // pad_enable
              | ((uint32_t)pad_i << 22)                     // pad_interval code
              | ((uint32_t)pad_a << 25);                    // pad_amount code
  g1[0] = (int)d0;
  g1[1] = (int)(((uint32_t)cols_halves & 0xFFFFu) << 16);   // tensor_dim0[15:0]
  g1[2] = (int)(((uint32_t)rows & 0xFFFFu) << 16);          // tensor_dim0 hi / tensor_dim1 lo
  g1[3] = (int)(((uint32_t)cols_halves & 0xFFFFu) << 16);   // tensor_dim1 hi / tile_dim0
  g1[4] = rows;                                             // tile_dim1 (tile_dim2 = 0)
  g1[5] = cols_halves;                                      // tensor_dim0_stride low
  g1[6] = 0;
  g1[7] = 0;
  i32x4 z4 = {0, 0, 0, 0};
#if __clang_major__ >= 23
  i32x8 z8 = {0, 0, 0, 0, 0, 0, 0, 0};
  __builtin_amdgcn_tensor_load_to_lds(g0, g1, z4, z4, z8, 0);
#else
  __builtin_amdgcn_tensor_load_to_lds(g0, g1, z4, z4, 0);
#endif
}

// --------------------------------------------------------------------------
// WMMA fragment loads as 2x ds_load_b128.
// A-pattern: lane holds k = {8*lh..8*lh+7} and {16+8*lh..16+8*lh+7}.
// B-pattern: same k split as A (row must be k-contiguous).
// --------------------------------------------------------------------------
__device__ __forceinline__ v16h load_frag_a(const _Float16* row, int lh) {
  union { v16h v; v8h h[2]; } u;
  u.h[0] = *(const v8h*)(row + (lh << 3));
  u.h[1] = *(const v8h*)(row + 16 + (lh << 3));
  return u.v;
}
__device__ __forceinline__ v16h load_frag_b(const _Float16* row, int lh) {
  union { v16h v; v8h h[2]; } u;
  u.h[0] = *(const v8h*)(row + (lh << 3));
  u.h[1] = *(const v8h*)(row + 16 + (lh << 3));
  return u.v;
}

// ---------------------------------------------------------------------------
// Generic GEMM: C[M][N] = A[M][K] * B[K][N], f32 in/out, f16 WMMA compute.
// Block tile 64x64, BK=32, 4 waves (2x2), each wave 32x32 (2x2 WMMA frags).
// B tile stored k-innermost (transposed) in LDS for b128 fragment loads.
// ---------------------------------------------------------------------------
__global__ __launch_bounds__(128)
void gemm_f16_kernel(const float* __restrict__ A, const float* __restrict__ Bm,
                     float* __restrict__ C, int M, int N, int K) {
  __shared__ alignas(16) _Float16 As[64][40];    // [m][k], 80B pitch
  __shared__ alignas(16) _Float16 BsT[64][40];   // [n][k], 80B pitch
  const int tid = threadIdx.x;
  const int lane = tid & 31, wave = tid >> 5;
  const int lm = lane & 15, lh = lane >> 4;
  const int wm = (wave >> 1) * 32, wn = (wave & 1) * 32;
  const int m0 = blockIdx.y * 64, n0 = blockIdx.x * 64;
  (void)M;
  v8f acc[2][2] = {};
  for (int k0 = 0; k0 < K; k0 += 32) {
    __syncthreads();
    // A tile: float4 loads, packed-half ds_store_b64.
    for (int idx = tid; idx < 64 * 8; idx += 128) {
      int r = idx >> 3, c4 = (idx & 7) << 2;
      const float4 f = *(const float4*)&A[(size_t)(m0 + r) * K + k0 + c4];
      v4h hv = {(_Float16)f.x, (_Float16)f.y, (_Float16)f.z, (_Float16)f.w};
      *(v4h*)&As[r][c4] = hv;
    }
    // B tile transposed: lanes sweep n (coalesced), 4 k-rows each.
    for (int idx = tid; idx < 64 * 8; idx += 128) {
      int n = idx & 63, kg = (idx >> 6) << 2;
      const float* bp = &Bm[(size_t)(k0 + kg) * N + n0 + n];
      v4h hv = {(_Float16)bp[0], (_Float16)bp[(size_t)N],
                (_Float16)bp[(size_t)2 * N], (_Float16)bp[(size_t)3 * N]};
      *(v4h*)&BsT[n][kg] = hv;
    }
    // Prefetch next k-tile into L2 while this one computes.
    if (k0 + 32 < K) {
      __builtin_prefetch(&A[(size_t)(m0 + (tid >> 1)) * K + k0 + 32], 0, 0);
      __builtin_prefetch(&Bm[(size_t)(k0 + 32 + (tid >> 2)) * N + n0], 0, 0);
    }
    __syncthreads();
    v16h af[2], bf[2];
#pragma unroll
    for (int t = 0; t < 2; ++t) {
      af[t] = load_frag_a(&As[wm + t * 16 + lm][0], lh);
      bf[t] = load_frag_b(&BsT[wn + t * 16 + lm][0], lh);
    }
#pragma unroll
    for (int tm = 0; tm < 2; ++tm)
#pragma unroll
      for (int tn = 0; tn < 2; ++tn)
        acc[tm][tn] = WMMA_F16(af[tm], bf[tn], acc[tm][tn]);
  }
#pragma unroll
  for (int tm = 0; tm < 2; ++tm)
#pragma unroll
    for (int tn = 0; tn < 2; ++tn)
#pragma unroll
      for (int r = 0; r < 8; ++r) {
        int row = m0 + wm + tm * 16 + r + 8 * lh;         // C frag: m=r+8*half
        int col = n0 + wn + tn * 16 + lm;
        C[(size_t)row * N + col] = acc[tm][tn][r];
      }
}

// ---------------------------------------------------------------------------
// RoPE + repack: Q[S][H*HD], K/V[S][KVH*HD] (f32) -> head-major f16 copies
// (+ post-rope f32 K for the k_hh gather).
// ---------------------------------------------------------------------------
__global__ void rope_pack_kernel(const float* __restrict__ Q, const float* __restrict__ K,
                                 const float* __restrict__ V,
                                 const float* __restrict__ cosT, const float* __restrict__ sinT,
                                 _Float16* __restrict__ qh, _Float16* __restrict__ kh,
                                 _Float16* __restrict__ vh, float* __restrict__ krf) {
  int idx = blockIdx.x * blockDim.x + threadIdx.x;
  if (idx >= Sn * Hn * HDn) return;
  int d = idx & (HDn - 1);
  int h = (idx >> 7) & (Hn - 1);
  int s = idx >> 12;
  float c = cosT[s * HDn + d], sn = sinT[s * HDn + d];
  size_t qb = (size_t)s * Dn + h * HDn;
  float qv = Q[qb + d];
  float qrot = (d < HDn / 2) ? -Q[qb + d + HDn / 2] : Q[qb + d - HDn / 2];
  qh[((size_t)h * Sn + s) * HDn + d] = (_Float16)(qv * c + qrot * sn);
  if (h < KVHn) {
    size_t kb = (size_t)s * (KVHn * HDn) + h * HDn;
    float kv_ = K[kb + d];
    float krot = (d < HDn / 2) ? -K[kb + d + HDn / 2] : K[kb + d - HDn / 2];
    float kr = kv_ * c + krot * sn;
    kh[((size_t)h * Sn + s) * HDn + d] = (_Float16)kr;
    krf[((size_t)h * Sn + s) * HDn + d] = kr;
    vh[((size_t)h * Sn + s) * HDn + d] = (_Float16)V[kb + d];
  }
}

// ---------------------------------------------------------------------------
// Flash attention forward: per (64-query tile, head). Online softmax, WMMA for
// QK^T and PV. Q async-staged, K staged by TDM (padded rows), V transposed.
// ---------------------------------------------------------------------------
__global__ __launch_bounds__(128)
void attn_fwd_kernel(const _Float16* __restrict__ qh, const _Float16* __restrict__ kh,
                     const _Float16* __restrict__ vh, float* __restrict__ o,
                     float* __restrict__ mws, float* __restrict__ lws) {
  __shared__ alignas(16) _Float16 Qs[64][136];    // [q][hd], 272B pitch
  __shared__ alignas(16) _Float16 Ks[32][136];    // [key][hd] (TDM-padded)
  __shared__ alignas(16) _Float16 VsT[128][40];   // [hd][key] (transposed)
  __shared__ alignas(16) _Float16 Ps[64][40];     // [q][key]
  const int h = blockIdx.y, hk = h / NREP;
  const int q0 = blockIdx.x * 64;
  const int tid = threadIdx.x, lane = tid & 31, wave = tid >> 5;
  const int lm = lane & 15, lh = lane >> 4;

  // Q tile: 64 rows x 128 halves = 1024 async b128 chunks.
  for (int idx = tid; idx < 64 * 16; idx += 128) {
    int r = idx >> 4, c8 = (idx & 15) << 3;
    async_copy_b128(&Qs[r][c8], qh + ((size_t)h * Sn + q0 + r) * HDn + c8);
  }
  wait_async_zero();
  __syncthreads();

  v16h aq[4];
#pragma unroll
  for (int kc = 0; kc < 4; ++kc)
    aq[kc] = load_frag_a(&Qs[wave * 16 + lm][kc * 32], lh);

  v8f oacc[8] = {};
  float mrow[8], lrow[8];
#pragma unroll
  for (int r = 0; r < 8; ++r) { mrow[r] = -1e30f; lrow[r] = 0.f; }
  const float scale = 0.08838834764831845f;   // 1/sqrt(128)
  const int nkt = (q0 >> 5) + 2;

  for (int kt = 0; kt < nkt; ++kt) {
    const int kbase = kt * 32;
    __syncthreads();
    // K tile via Tensor Data Mover: 32x128 f16, pad 4 DW after every 64 DW
    // (row 256B -> LDS pitch 272B, matching Ks[32][136]).
    if (wave == 0)
      tdm_load_tile_f16(&Ks[0][0], kh + ((size_t)hk * Sn + kbase) * HDn,
                        /*rows=*/32, /*cols=*/HDn, /*pad_i=*/5, /*pad_a=*/3);
    // V tile: transpose into [hd][key] so PV B-frags are k-contiguous.
    for (int idx = tid; idx < 32 * 16; idx += 128) {
      int key = idx >> 4, c8 = (idx & 15) << 3;
      v8h vv = *(const v8h*)(vh + ((size_t)hk * Sn + kbase + key) * HDn + c8);
#pragma unroll
      for (int j = 0; j < 8; ++j) VsT[c8 + j][key] = vv[j];
    }
    if (wave == 0) __builtin_amdgcn_s_wait_tensorcnt(0);
    __syncthreads();

    v8f sf[2] = {};
#pragma unroll
    for (int nt = 0; nt < 2; ++nt)
#pragma unroll
      for (int kc = 0; kc < 4; ++kc) {
        v16h bk = load_frag_b(&Ks[nt * 16 + lm][kc * 32], lh);  // n=key, k=hd
        sf[nt] = WMMA_F16(aq[kc], bk, sf[nt]);
      }

    float p0v[8], p1v[8], rmaxn[8];
#pragma unroll
    for (int r = 0; r < 8; ++r) {
      int rowg = q0 + wave * 16 + r + 8 * lh;
      float v0 = sf[0][r] * scale, v1 = sf[1][r] * scale;
      if (kbase + lm > rowg) v0 = -1e9f;
      if (kbase + 16 + lm > rowg) v1 = -1e9f;
      p0v[r] = v0; p1v[r] = v1;
      float t = fmaxf(v0, v1);
      t = fmaxf(t, __shfl_xor(t, 1, 32));
      t = fmaxf(t, __shfl_xor(t, 2, 32));
      t = fmaxf(t, __shfl_xor(t, 4, 32));
      t = fmaxf(t, __shfl_xor(t, 8, 32));
      rmaxn[r] = t;
    }
#pragma unroll
    for (int r = 0; r < 8; ++r) {
      float mnew = fmaxf(mrow[r], rmaxn[r]);
      float sc = __expf(mrow[r] - mnew);
      float e0 = __expf(p0v[r] - mnew);
      float e1 = __expf(p1v[r] - mnew);
      float rs = e0 + e1;
      rs += __shfl_xor(rs, 1, 32);
      rs += __shfl_xor(rs, 2, 32);
      rs += __shfl_xor(rs, 4, 32);
      rs += __shfl_xor(rs, 8, 32);
      lrow[r] = lrow[r] * sc + rs;
      mrow[r] = mnew;
#pragma unroll
      for (int nf = 0; nf < 8; ++nf) oacc[nf][r] *= sc;
      Ps[wave * 16 + r + 8 * lh][lm] = (_Float16)e0;
      Ps[wave * 16 + r + 8 * lh][16 + lm] = (_Float16)e1;
    }

    v16h ap = load_frag_a(&Ps[wave * 16 + lm][0], lh);
#pragma unroll
    for (int nf = 0; nf < 8; ++nf) {
      v16h bv = load_frag_b(&VsT[nf * 16 + lm][0], lh);  // n=hd, k=key
      oacc[nf] = WMMA_F16(ap, bv, oacc[nf]);
    }
  }

#pragma unroll
  for (int r = 0; r < 8; ++r) {
    int rowg = q0 + wave * 16 + r + 8 * lh;
    float inv = 1.0f / lrow[r];
#pragma unroll
    for (int nf = 0; nf < 8; ++nf)
      o[(size_t)rowg * Dn + h * HDn + nf * 16 + lm] = oacc[nf][r] * inv;
    if (lm == 0) { mws[h * Sn + rowg] = mrow[r]; lws[h * Sn + rowg] = lrow[r]; }
  }
}

// ---------------------------------------------------------------------------
// hh pass: recompute score tiles (WMMA), form final probs with saved m,l,
// column-sum and atomically accumulate hh[kvh][key] (mean over head group).
// ---------------------------------------------------------------------------
__global__ __launch_bounds__(128)
void hh_accum_kernel(const _Float16* __restrict__ qh, const _Float16* __restrict__ kh,
                     const float* __restrict__ mws, const float* __restrict__ lws,
                     float* __restrict__ hh) {
  __shared__ alignas(16) _Float16 Qs[64][136];
  __shared__ alignas(16) _Float16 Ks[32][136];
  const int h = blockIdx.y, hk = h / NREP;
  const int q0 = blockIdx.x * 64;
  const int tid = threadIdx.x, lane = tid & 31, wave = tid >> 5;
  const int lm = lane & 15, lh = lane >> 4;

  for (int idx = tid; idx < 64 * 16; idx += 128) {
    int r = idx >> 4, c8 = (idx & 15) << 3;
    async_copy_b128(&Qs[r][c8], qh + ((size_t)h * Sn + q0 + r) * HDn + c8);
  }
  wait_async_zero();
  __syncthreads();
  v16h aq[4];
#pragma unroll
  for (int kc = 0; kc < 4; ++kc)
    aq[kc] = load_frag_a(&Qs[wave * 16 + lm][kc * 32], lh);

  float mr[8], li[8];
#pragma unroll
  for (int r = 0; r < 8; ++r) {
    int rowg = q0 + wave * 16 + r + 8 * lh;
    mr[r] = mws[h * Sn + rowg];
    li[r] = 1.0f / lws[h * Sn + rowg];
  }
  const float scale = 0.08838834764831845f;
  const int nkt = (q0 >> 5) + 2;
  for (int kt = 0; kt < nkt; ++kt) {
    const int kbase = kt * 32;
    __syncthreads();
    if (wave == 0) {
      tdm_load_tile_f16(&Ks[0][0], kh + ((size_t)hk * Sn + kbase) * HDn,
                        /*rows=*/32, /*cols=*/HDn, /*pad_i=*/5, /*pad_a=*/3);
      __builtin_amdgcn_s_wait_tensorcnt(0);
    }
    __syncthreads();
    v8f sf[2] = {};
#pragma unroll
    for (int nt = 0; nt < 2; ++nt)
#pragma unroll
      for (int kc = 0; kc < 4; ++kc) {
        v16h bk = load_frag_b(&Ks[nt * 16 + lm][kc * 32], lh);
        sf[nt] = WMMA_F16(aq[kc], bk, sf[nt]);
      }
#pragma unroll
    for (int nt = 0; nt < 2; ++nt) {
      int colg = kbase + nt * 16 + lm;
      float csum = 0.f;
#pragma unroll
      for (int r = 0; r < 8; ++r) {
        int rowg = q0 + wave * 16 + r + 8 * lh;
        if (colg <= rowg)
          csum += __expf(sf[nt][r] * scale - mr[r]) * li[r];
      }
      csum += __shfl_xor(csum, 16, 32);          // add the other half's 8 rows
      if (lh == 0)
        atomicAdd(&hh[hk * Sn + colg], csum * (1.0f / NREP));
    }
  }
}

// ---------------------------------------------------------------------------
// Top-HH selection per KV head over columns [0, S-RECENT), rank-sorted
// ascending, recent window appended.
// ---------------------------------------------------------------------------
__global__ __launch_bounds__(256)
void topk_kernel(const float* __restrict__ hh, int* __restrict__ keep) {
  const int kv = blockIdx.x;
  __shared__ float vals[SELN];
  __shared__ float rmax[256];
  __shared__ int   ridx[256];
  __shared__ int   sel[HHn];
  const int tid = threadIdx.x;
  for (int i = tid; i < SELN; i += 256) vals[i] = hh[kv * Sn + i];
  __syncthreads();
  for (int t = 0; t < HHn; ++t) {
    float bm = -1e30f; int bi = SELN;
    for (int i = tid; i < SELN; i += 256) {
      float v = vals[i];
      if (v > bm) { bm = v; bi = i; }
    }
    rmax[tid] = bm; ridx[tid] = bi;
    __syncthreads();
    for (int s = 128; s > 0; s >>= 1) {
      if (tid < s) {
        bool take = (rmax[tid + s] > rmax[tid]) ||
                    (rmax[tid + s] == rmax[tid] && ridx[tid + s] < ridx[tid]);
        if (take) { rmax[tid] = rmax[tid + s]; ridx[tid] = ridx[tid + s]; }
      }
      __syncthreads();
    }
    if (tid == 0) { sel[t] = ridx[0]; vals[ridx[0]] = -1e30f; }
    __syncthreads();
  }
  if (tid < HHn) {
    int my = sel[tid], rank = 0;
    for (int j = 0; j < HHn; ++j) rank += (sel[j] < my) ? 1 : 0;
    keep[kv * NKEEP + rank] = my;
  }
  for (int i = tid; i < RECENTn; i += 256)
    keep[kv * NKEEP + HHn + i] = SELN + i;
}

// ---------------------------------------------------------------------------
// Gather kept K/V rows and hh values into d_out.
// ---------------------------------------------------------------------------
__global__ void gather_kernel(const float* __restrict__ krf, const float* __restrict__ V,
                              const float* __restrict__ hh, const int* __restrict__ keep,
                              float* __restrict__ outk, float* __restrict__ outv,
                              float* __restrict__ outhh) {
  int idx = blockIdx.x * blockDim.x + threadIdx.x;
  if (idx >= KVHn * NKEEP * HDn) return;
  int d = idx & (HDn - 1);
  int j = (idx >> 7) % NKEEP;
  int kv = idx / (NKEEP * HDn);
  int key = keep[kv * NKEEP + j];
  outk[idx] = krf[((size_t)kv * Sn + key) * HDn + d];
  outv[idx] = V[(size_t)key * (KVHn * HDn) + kv * HDn + d];
  if (d == 0) outhh[kv * NKEEP + j] = hh[kv * Sn + key];
}

// ---------------------------------------------------------------------------
extern "C" void kernel_launch(void* const* d_in, const int* in_sizes, int n_in,
                              void* d_out, int out_size, void* d_ws, size_t ws_size,
                              hipStream_t stream) {
  (void)in_sizes; (void)n_in; (void)out_size; (void)ws_size;
  const float* x    = (const float*)d_in[0];
  const float* Wq   = (const float*)d_in[1];
  const float* Wk   = (const float*)d_in[2];
  const float* Wv   = (const float*)d_in[3];
  const float* Wo   = (const float*)d_in[4];
  const float* cosT = (const float*)d_in[5];
  const float* sinT = (const float*)d_in[6];
  // d_in[7] = position_ids == iota, not needed.

  float* out = (float*)d_out;
  float* outk  = out + (size_t)Sn * Dn;
  float* outv  = outk + (size_t)KVHn * NKEEP * HDn;
  float* outhh = outv + (size_t)KVHn * NKEEP * HDn;

  char* ws = (char*)d_ws;
  size_t off = 0;
  auto carve = [&](size_t bytes) -> void* {
    void* p = ws + off;
    off = (off + bytes + 255) & ~(size_t)255;
    return p;
  };
  float*    Qf  = (float*)carve((size_t)Sn * Dn * 4);
  float*    Kf  = (float*)carve((size_t)Sn * KVHn * HDn * 4);
  float*    Vf  = (float*)carve((size_t)Sn * KVHn * HDn * 4);
  float*    Of  = (float*)carve((size_t)Sn * Dn * 4);
  _Float16* qhp = (_Float16*)carve((size_t)Hn * Sn * HDn * 2);
  _Float16* khp = (_Float16*)carve((size_t)KVHn * Sn * HDn * 2);
  _Float16* vhp = (_Float16*)carve((size_t)KVHn * Sn * HDn * 2);
  float*    krf = (float*)carve((size_t)KVHn * Sn * HDn * 4);
  float*    mws = (float*)carve((size_t)Hn * Sn * 4);
  float*    lws = (float*)carve((size_t)Hn * Sn * 4);
  float*    hhw = (float*)carve((size_t)KVHn * Sn * 4);
  int*      keep = (int*)carve((size_t)KVHn * NKEEP * 4);

  // 1) QKV projections (f16 WMMA GEMMs)
  gemm_f16_kernel<<<dim3(Dn / 64, Sn / 64), 128, 0, stream>>>(x, Wq, Qf, Sn, Dn, Dn);
  gemm_f16_kernel<<<dim3((KVHn * HDn) / 64, Sn / 64), 128, 0, stream>>>(x, Wk, Kf, Sn, KVHn * HDn, Dn);
  gemm_f16_kernel<<<dim3((KVHn * HDn) / 64, Sn / 64), 128, 0, stream>>>(x, Wv, Vf, Sn, KVHn * HDn, Dn);

  // 2) RoPE + head-major f16 repack
  {
    int total = Sn * Hn * HDn;
    rope_pack_kernel<<<(total + 255) / 256, 256, 0, stream>>>(Qf, Kf, Vf, cosT, sinT,
                                                              qhp, khp, vhp, krf);
  }

  // 3) Flash attention forward (WMMA; TDM K-staging, async Q-staging)
  attn_fwd_kernel<<<dim3(Sn / 64, Hn), 128, 0, stream>>>(qhp, khp, vhp, Of, mws, lws);

  // 4) hh column sums
  hipMemsetAsync(hhw, 0, (size_t)KVHn * Sn * 4, stream);
  hh_accum_kernel<<<dim3(Sn / 64, Hn), 128, 0, stream>>>(qhp, khp, mws, lws, hhw);

  // 5) Heavy-hitter top-k + gather into d_out
  topk_kernel<<<KVHn, 256, 0, stream>>>(hhw, keep);
  {
    int total = KVHn * NKEEP * HDn;
    gather_kernel<<<(total + 255) / 256, 256, 0, stream>>>(krf, Vf, hhw, keep,
                                                           outk, outv, outhh);
  }

  // 6) Output projection into d_out[0 .. S*D)
  gemm_f16_kernel<<<dim3(Dn / 64, Sn / 64), 128, 0, stream>>>(Of, Wo, out, Sn, Dn, Dn);
}
